// LSTMModel_14027363188766
// MI455X (gfx1250) — compile-verified, matrix operand order checked by reading, not run on hardware
//
#include <hip/hip_runtime.h>
#include <hip/hip_bf16.h>

// ---------------------------------------------------------------------------
// Fused 2-layer LSTM (B=1024, T=512, H=64, D_IN=40) + final FC for gfx1250.
// One workgroup = 16 batch rows, persistent over all 512 timesteps.
// 16 waves/WG; wave w computes gate columns [16w,16w+16) with
// v_wmma_f32_16x16x32_bf16 (two independent 2-deep chains per phase).
// Weights live in registers (B-fragments), activations/gate exchange go
// through LDS, cell state in thread registers. Activations use
// v_exp_f32 + v_rcp_f32 (TRANS ops, co-issue with WMMA).
// ---------------------------------------------------------------------------

typedef __attribute__((ext_vector_type(16))) __bf16 v16bf;
typedef __attribute__((ext_vector_type(8)))  float  v8f;

#define SEQ_T   512
#define BATCH   1024
#define HID     64
#define GATES   256   // 4*H
#define DIN_RAW 40    // 32 x + 8 u
#define DIN_PAD 64    // padded K for WMMA

__device__ __forceinline__ unsigned short f2bf(float f) {
  union { float f; unsigned u; } v; v.f = f;
  unsigned r = v.u + 0x7FFFu + ((v.u >> 16) & 1u);   // round-to-nearest-even
  return (unsigned short)(r >> 16);
}
__device__ __forceinline__ float bf2f(unsigned short b) {
  union { unsigned u; float f; } v; v.u = ((unsigned)b) << 16;
  return v.f;
}

// sigmoid via v_exp_f32 + v_rcp_f32 (no IEEE divide sequence)
__device__ __forceinline__ float fast_sigmoid(float x) {
  return __builtin_amdgcn_rcpf(1.f + __expf(-x));
}
__device__ __forceinline__ float fast_tanh(float x) {
  return 2.f * __builtin_amdgcn_rcpf(1.f + __expf(-2.f * x)) - 1.f;
}

union FragAB { v16bf v; uint4 q[2]; };

// A-fragment (16x32 bf16) from an LDS tile stored row-major [16][64] bf16.
// Lane L: row M = L%16; regs 0-3 hold K = kb*32 + (L/16)*8 + 0..7,
//         regs 4-7 hold K = kb*32 + 16 + (L/16)*8 + 0..7  (ISA 7.12.2).
__device__ __forceinline__ FragAB load_a_frag(const unsigned short* A, int kb, int lane) {
  const int m   = lane & 15;
  const int khi = (lane >> 4) << 3;
  const unsigned short* p = A + m * DIN_PAD + kb * 32 + khi;
  FragAB f;
  f.q[0] = *reinterpret_cast<const uint4*>(p);
  f.q[1] = *reinterpret_cast<const uint4*>(p + 16);
  return f;
}

// B-fragment (32x16 bf16) for column n from row-major bf16 weights W[n][k]
// (ld = 64). Lane L covers column n; same K striping as A.
__device__ __forceinline__ FragAB load_b_frag(const unsigned short* W, int n, int kb, int lane) {
  const int khi = (lane >> 4) << 3;
  const unsigned short* p = W + n * DIN_PAD + kb * 32 + khi;
  FragAB f;
  f.q[0] = *reinterpret_cast<const uint4*>(p);
  f.q[1] = *reinterpret_cast<const uint4*>(p + 16);
  return f;
}

__device__ __forceinline__ v8f wmma_bf16(const FragAB& a, const FragAB& b, v8f c) {
  return __builtin_amdgcn_wmma_f32_16x16x32_bf16(
      /*neg_a=*/false, a.v, /*neg_b=*/false, b.v,
      /*c_mod=*/(short)0, c, /*reuse_a=*/false, /*reuse_b=*/false);
}

// ---------------------------------------------------------------------------
// Prologue: pack fp32 weights into padded bf16 [256][64] matrices in ws.
// ---------------------------------------------------------------------------
__global__ void pack_weights(const float* __restrict__ W_ih0,
                             const float* __restrict__ W_hh0,
                             const float* __restrict__ W_ih1,
                             const float* __restrict__ W_hh1,
                             unsigned short* __restrict__ Wbf) {
  int i = blockIdx.x * blockDim.x + threadIdx.x;   // 0 .. 16383
  if (i >= GATES * DIN_PAD) return;
  int n = i >> 6, k = i & 63;
  Wbf[i]                       = (k < DIN_RAW) ? f2bf(W_ih0[n * DIN_RAW + k]) : (unsigned short)0;
  Wbf[GATES * DIN_PAD     + i] = f2bf(W_hh0[n * HID + k]);
  Wbf[GATES * DIN_PAD * 2 + i] = f2bf(W_ih1[n * HID + k]);
  Wbf[GATES * DIN_PAD * 3 + i] = f2bf(W_hh1[n * HID + k]);
}

// ---------------------------------------------------------------------------
// Fused persistent LSTM kernel. Grid = 64 blocks (16 batch rows each),
// block = 512 threads = 16 wave32s.
// ---------------------------------------------------------------------------
__global__ __launch_bounds__(512)
void lstm_fused(const float* __restrict__ x_seq,   // (B,T,32)
                const float* __restrict__ u_seq,   // (B,T,8)
                const float* __restrict__ b_ih0, const float* __restrict__ b_hh0,
                const float* __restrict__ b_ih1, const float* __restrict__ b_hh1,
                const float* __restrict__ W_fc,    // (32,64) fp32
                const float* __restrict__ b_fc,    // (32,)
                const unsigned short* __restrict__ Wbf,
                float* __restrict__ out)           // (B,32)
{
  __shared__ __align__(16) unsigned short A0[16 * DIN_PAD];     // layer-0 input tile (bf16)
  __shared__ __align__(16) unsigned short h0buf[16 * DIN_PAD];  // layer-0 hidden (bf16)
  __shared__ __align__(16) unsigned short h1buf[16 * DIN_PAD];  // layer-1 hidden (bf16)
  __shared__ __align__(16) float gb[16][GATES];                 // activated gates (fp32)

  const int tid  = threadIdx.x;
  const int lane = tid & 31;
  const int w    = tid >> 5;            // wave id 0..15
  const int wn   = w << 4;              // gate-column tile base
  const int bm0  = blockIdx.x << 4;     // first batch row of this WG

  // ---- weight B-fragments: resident in registers for all 512 steps --------
  const unsigned short* Wih0 = Wbf;
  const unsigned short* Whh0 = Wbf + GATES * DIN_PAD;
  const unsigned short* Wih1 = Wbf + GATES * DIN_PAD * 2;
  const unsigned short* Whh1 = Wbf + GATES * DIN_PAD * 3;
  const int ncol = wn + (lane & 15);
  FragAB fih0[2], fhh0[2], fih1[2], fhh1[2];
#pragma unroll
  for (int kb = 0; kb < 2; ++kb) {
    fih0[kb] = load_b_frag(Wih0, ncol, kb, lane);
    fhh0[kb] = load_b_frag(Whh0, ncol, kb, lane);
    fih1[kb] = load_b_frag(Wih1, ncol, kb, lane);
    fhh1[kb] = load_b_frag(Whh1, ncol, kb, lane);
  }
  const float bias0 = b_ih0[ncol] + b_hh0[ncol];   // per-lane column bias
  const float bias1 = b_ih1[ncol] + b_hh1[ncol];
  // gate activation: waves 8..11 hold the 'g' gate (tanh); others sigmoid.
  // tanh(x) = 2*sigmoid(2x)-1  -> branchless cndmask selection.
  const bool  is_g   = ((w >> 2) == 2);
  const float in_sc  = is_g ? 2.f : 1.f;
  const float out_sc = is_g ? 2.f : 1.f;
  const float out_of = is_g ? -1.f : 0.f;

  // D-fragment scatter coordinates (reg r, lane L) -> (row r+mb, col nc)
  const int mb = (lane >> 4) << 3;
  const int nc = ncol;

  // ---- init LDS state + zero K-padding of A0 ------------------------------
  for (int i = tid; i < 16 * DIN_PAD; i += 512) { h0buf[i] = 0; h1buf[i] = 0; }
  for (int i = tid; i < 16 * (DIN_PAD - DIN_RAW); i += 512) {
    int r = i / (DIN_PAD - DIN_RAW), c = i % (DIN_PAD - DIN_RAW);
    A0[r * DIN_PAD + DIN_RAW + c] = 0;
  }

  // cell state: thread owns 2 consecutive (m,j) slots per layer, fp32
  float c0[2] = {0.f, 0.f}, c1[2] = {0.f, 0.f};

  // elementwise-phase coordinates: row m, columns j, j+1 (j even)
  const int em = tid >> 5;
  const int ej = (tid & 31) << 1;

  // input staging indices for this thread
  const int xr = tid >> 5, xc = tid & 31;                 // 16x32 x-tile
  const size_t xrow = ((size_t)(bm0 + xr) * SEQ_T);
  const int ur = tid >> 3, uc = tid & 7;                  // 16x8 u-tile (tid<128)
  const size_t urow = ((size_t)(bm0 + ur) * SEQ_T);
  const int xoff = xr * DIN_PAD + xc;
  const int uoff = ur * DIN_PAD + 32 + uc;

  // stage t = 0 tile
  A0[xoff] = f2bf(x_seq[xrow * 32 + xc]);
  if (tid < 128) A0[uoff] = f2bf(u_seq[urow * 8 + uc]);
  __syncthreads();

  for (int t = 0; t < SEQ_T; ++t) {
    // ---- prefetch next timestep's inputs into registers (no wait here) ----
    float xn = 0.f, un = 0.f;
    if (t + 1 < SEQ_T) {
      xn = x_seq[(xrow + t + 1) * 32 + xc];
      if (tid < 128) un = u_seq[(urow + t + 1) * 8 + uc];
      if (t + 2 < SEQ_T)
        __builtin_prefetch(&x_seq[(xrow + t + 2) * 32 + xc], 0, 1);
    }

    // ---- layer 0 gates: two independent 2-deep WMMA chains ----------------
    {
      v8f accA = {}, accH = {};
      accA = wmma_bf16(load_a_frag(A0, 0, lane), fih0[0], accA);
      accH = wmma_bf16(load_a_frag(h0buf, 0, lane), fhh0[0], accH);
      accA = wmma_bf16(load_a_frag(A0, 1, lane), fih0[1], accA);
      accH = wmma_bf16(load_a_frag(h0buf, 1, lane), fhh0[1], accH);
#pragma unroll
      for (int r = 0; r < 8; ++r) {
        float g = (accA[r] + accH[r]) + bias0;
        gb[r + mb][nc] = out_sc * fast_sigmoid(in_sc * g) + out_of;
      }
    }
    __syncthreads();

    // ---- layer 0 elementwise: c0,h0 update (packed b64 reads, b32 write) --
    {
      float2 iv = *reinterpret_cast<const float2*>(&gb[em][ej]);
      float2 fv = *reinterpret_cast<const float2*>(&gb[em][64 + ej]);
      float2 gv = *reinterpret_cast<const float2*>(&gb[em][128 + ej]);
      float2 ov = *reinterpret_cast<const float2*>(&gb[em][192 + ej]);
      c0[0] = fv.x * c0[0] + iv.x * gv.x;
      c0[1] = fv.y * c0[1] + iv.y * gv.y;
      unsigned pk = (unsigned)f2bf(ov.x * fast_tanh(c0[0])) |
                    ((unsigned)f2bf(ov.y * fast_tanh(c0[1])) << 16);
      *reinterpret_cast<unsigned*>(&h0buf[em * DIN_PAD + ej]) = pk;
    }
    __syncthreads();

    // ---- layer 1 gates ----------------------------------------------------
    {
      v8f accA = {}, accH = {};
      accA = wmma_bf16(load_a_frag(h0buf, 0, lane), fih1[0], accA);
      accH = wmma_bf16(load_a_frag(h1buf, 0, lane), fhh1[0], accH);
      accA = wmma_bf16(load_a_frag(h0buf, 1, lane), fih1[1], accA);
      accH = wmma_bf16(load_a_frag(h1buf, 1, lane), fhh1[1], accH);
#pragma unroll
      for (int r = 0; r < 8; ++r) {
        float g = (accA[r] + accH[r]) + bias1;
        gb[r + mb][nc] = out_sc * fast_sigmoid(in_sc * g) + out_of;
      }
    }
    __syncthreads();

    // ---- layer 1 elementwise + commit next A0 tile ------------------------
    {
      float2 iv = *reinterpret_cast<const float2*>(&gb[em][ej]);
      float2 fv = *reinterpret_cast<const float2*>(&gb[em][64 + ej]);
      float2 gv = *reinterpret_cast<const float2*>(&gb[em][128 + ej]);
      float2 ov = *reinterpret_cast<const float2*>(&gb[em][192 + ej]);
      c1[0] = fv.x * c1[0] + iv.x * gv.x;
      c1[1] = fv.y * c1[1] + iv.y * gv.y;
      unsigned pk = (unsigned)f2bf(ov.x * fast_tanh(c1[0])) |
                    ((unsigned)f2bf(ov.y * fast_tanh(c1[1])) << 16);
      *reinterpret_cast<unsigned*>(&h1buf[em * DIN_PAD + ej]) = pk;
    }
    if (t + 1 < SEQ_T) {
      // A0's last reader was the layer-0 WMMA phase (3 barriers ago): safe.
      A0[xoff] = f2bf(xn);
      if (tid < 128) A0[uoff] = f2bf(un);
    }
    __syncthreads();
  }

  // ---- final FC: out[bm0+m][n] = b_fc[n] + sum_j h1[m][j] * W_fc[n][j] ----
  {
    const int m = tid >> 5;     // 0..15
    const int n = tid & 31;     // 0..31
    float s = b_fc[n];
#pragma unroll
    for (int j = 0; j < HID; ++j)
      s += bf2f(h1buf[m * DIN_PAD + j]) * W_fc[n * HID + j];
    out[(size_t)(bm0 + m) * 32 + n] = s;
  }
}

// ---------------------------------------------------------------------------
extern "C" void kernel_launch(void* const* d_in, const int* in_sizes, int n_in,
                              void* d_out, int out_size, void* d_ws, size_t ws_size,
                              hipStream_t stream) {
  const float* x_seq = (const float*)d_in[0];
  const float* u_seq = (const float*)d_in[1];
  const float* W_ih0 = (const float*)d_in[2];
  const float* W_hh0 = (const float*)d_in[3];
  const float* b_ih0 = (const float*)d_in[4];
  const float* b_hh0 = (const float*)d_in[5];
  const float* W_ih1 = (const float*)d_in[6];
  const float* W_hh1 = (const float*)d_in[7];
  const float* b_ih1 = (const float*)d_in[8];
  const float* b_hh1 = (const float*)d_in[9];
  const float* W_fc  = (const float*)d_in[10];
  const float* b_fc  = (const float*)d_in[11];
  unsigned short* Wbf = (unsigned short*)d_ws;     // 128 KB of bf16 weights
  float* out = (float*)d_out;

  pack_weights<<<64, 256, 0, stream>>>(W_ih0, W_hh0, W_ih1, W_hh1, Wbf);
  lstm_fused<<<BATCH / 16, 512, 0, stream>>>(x_seq, u_seq, b_ih0, b_hh0,
                                             b_ih1, b_hh1, W_fc, b_fc, Wbf, out);
}
